// BubbleformerAxialAttentionBlock_6622839571045
// MI455X (gfx1250) — matrix operand order repaired and checked by reading, not compiled, not run on hardware
//
#include <hip/hip_runtime.h>
#include <math.h>
#include <stdint.h>

// ---------------------------------------------------------------------------
// Bubbleformer axial attention block for gfx1250 (MI455X), wave32 + WMMA.
// GEMMs/attention use v_wmma_f32_16x16x32_bf16 with LDS tiles pre-swizzled in
// fragment layout (2x ds_load_b128 per operand). Rel-bias staged via
// global_load_async_to_lds (ASYNCcnt) in the attention kernel.
// ---------------------------------------------------------------------------

typedef __attribute__((ext_vector_type(16))) __bf16 v16bf;
typedef __attribute__((ext_vector_type(4)))  __bf16 v4bf;
typedef __attribute__((ext_vector_type(8)))  float  v8f;

#define CEMB   768
#define NHEADS 12
#define HD     64
#define NB     8
#define SPAT   4096
#define SIDE   64

#define TILE_M 64
#define TILE_N 64
#define TILE_K 32

#define BMODE_F32_NORM 0
#define BMODE_F32      1
#define BMODE_BF16     2

#define EP_BF16_BIAS      0
#define EP_F32_BIAS       1
#define EP_BF16_BIAS_GELU 2

__device__ __forceinline__ float gelu_exact(float x) {
    return 0.5f * x * (1.0f + erff(x * 0.70710678118654752440f));
}

// ---- fragment-layout LDS helpers -----------------------------------------
__device__ __forceinline__ v16bf loadFrag(const __bf16* p, int ti, int lane) {
    return *(const v16bf*)(p + (ti << 9) + (lane << 4));
}
__device__ __forceinline__ int fragAoff(int m, int k) {
    return ((m >> 4) << 9) + (((((k >> 3) & 1) << 4) | (m & 15)) << 4)
         + ((k & 7) | ((k & 16) >> 1));
}
__device__ __forceinline__ int fragBoff(int n, int k) {
    return ((n >> 4) << 9) + (((((k >> 3) & 1) << 4) | (n & 15)) << 4)
         + ((k & 7) | ((k & 16) >> 1));
}
__device__ __forceinline__ v4bf pack4(float a, float b, float c, float d) {
    v4bf v; v[0] = (__bf16)a; v[1] = (__bf16)b; v[2] = (__bf16)c; v[3] = (__bf16)d;
    return v;
}

// ---------------------------------------------------------------------------
// Tiled bf16 WMMA GEMM: Out[b][m][n] = sum_k A[m][k]*B[b][k][n] (+ epilogue).
// Fully specialized on (TRANSA, BMODE, EPMODE); double-buffered fragment-layout
// LDS; per-thread pointer-bump staging; peeled final iteration.
// ---------------------------------------------------------------------------
template<int TRANSA, int BMODE, int EPMODE>
__global__ __launch_bounds__(256)
void gemm_wmma_kernel(const float* __restrict__ A, int lda,
                      const void* __restrict__ Bg,
                      const float* __restrict__ nscale, const float* __restrict__ nshift,
                      const float* __restrict__ bias,
                      void* __restrict__ Out, int M, int N, int K)
{
    const int bz = blockIdx.z;
    const int m0 = blockIdx.y * TILE_M;
    const int n0 = blockIdx.x * TILE_N;
    const int tid = threadIdx.x, lane = tid & 31, wave = tid >> 5;

    __shared__ __align__(32) __bf16 AsF[2][2048];
    __shared__ __align__(32) __bf16 BsF[2][2048];

    // ---- per-thread staging pointers (advanced by constant strides) ----
    const float* ap0;
    const float* ap1;
    size_t aStep;
    int aOff0, aOff1;
    if (TRANSA == 0) {
        int m = tid >> 3, kq = (tid & 7) * 4;
        ap0 = A + (size_t)(m0 + m) * lda + kq;
        ap1 = ap0 + (size_t)32 * lda;
        aOff0 = fragAoff(m, kq);
        aOff1 = fragAoff(m + 32, kq);
        aStep = TILE_K;
    } else {
        int m = tid & 63, kq = (tid >> 6) * 4;
        ap0 = A + (size_t)kq * lda + (m0 + m);
        ap1 = ap0 + (size_t)16 * lda;
        aOff0 = fragAoff(m, kq);
        aOff1 = fragAoff(m, kq + 16);
        aStep = (size_t)TILE_K * lda;
    }
    const int nB = tid & 63, kqB = (tid >> 6) * 4;
    const float*  bpf = (const float*)Bg  + (size_t)bz * K * N + (size_t)kqB * N + n0 + nB;
    const __bf16* bph = (const __bf16*)Bg + (size_t)bz * K * N + (size_t)kqB * N + n0 + nB;
    const float* np = nscale ? nscale + (size_t)bz * K + kqB : nullptr;
    const float* sp = nshift ? nshift + (size_t)bz * K + kqB : nullptr;
    const int bOff0 = fragBoff(nB, kqB);
    const int bOff1 = fragBoff(nB, kqB + 16);

    auto stage = [&](int buf) {
        __bf16* Ad = AsF[buf];
        __bf16* Bd = BsF[buf];
        if (TRANSA == 0) {
            float4 f0 = *(const float4*)ap0;
            float4 f1 = *(const float4*)ap1;
            *(v4bf*)(Ad + aOff0) = pack4(f0.x, f0.y, f0.z, f0.w);
            *(v4bf*)(Ad + aOff1) = pack4(f1.x, f1.y, f1.z, f1.w);
        } else {
            *(v4bf*)(Ad + aOff0) = pack4(ap0[0], ap0[(size_t)lda],
                                         ap0[(size_t)2 * lda], ap0[(size_t)3 * lda]);
            *(v4bf*)(Ad + aOff1) = pack4(ap1[0], ap1[(size_t)lda],
                                         ap1[(size_t)2 * lda], ap1[(size_t)3 * lda]);
        }
        ap0 += aStep; ap1 += aStep;

        float f[8];
        if (BMODE == BMODE_BF16) {
#pragma unroll
            for (int j = 0; j < 4; ++j) f[j]     = (float)bph[(size_t)j * N];
#pragma unroll
            for (int j = 0; j < 4; ++j) f[4 + j] = (float)bph[(size_t)(16 + j) * N];
            bph += (size_t)TILE_K * N;
            __builtin_prefetch(bph + (size_t)TILE_K * N, 0, 1);
        } else {
#pragma unroll
            for (int j = 0; j < 4; ++j) f[j]     = bpf[(size_t)j * N];
#pragma unroll
            for (int j = 0; j < 4; ++j) f[4 + j] = bpf[(size_t)(16 + j) * N];
            bpf += (size_t)TILE_K * N;
            __builtin_prefetch(bpf + (size_t)TILE_K * N, 0, 1);
        }
        if (BMODE == BMODE_F32_NORM) {
            float4 sa0 = *(const float4*)np;
            float4 sa1 = *(const float4*)(np + 16);
            float4 sb0 = *(const float4*)sp;
            float4 sb1 = *(const float4*)(sp + 16);
            f[0] = f[0] * sa0.x + sb0.x; f[1] = f[1] * sa0.y + sb0.y;
            f[2] = f[2] * sa0.z + sb0.z; f[3] = f[3] * sa0.w + sb0.w;
            f[4] = f[4] * sa1.x + sb1.x; f[5] = f[5] * sa1.y + sb1.y;
            f[6] = f[6] * sa1.z + sb1.z; f[7] = f[7] * sa1.w + sb1.w;
            np += TILE_K; sp += TILE_K;
        }
        *(v4bf*)(Bd + bOff0) = pack4(f[0], f[1], f[2], f[3]);
        *(v4bf*)(Bd + bOff1) = pack4(f[4], f[5], f[6], f[7]);
    };

    v8f acc0 = {}, acc1 = {};
    const int mi0 = wave >> 2, ni = wave & 3, mi1 = mi0 + 2;
    auto compute = [&](int buf) {
        v16bf a0 = loadFrag(AsF[buf], mi0, lane);
        v16bf a1 = loadFrag(AsF[buf], mi1, lane);
        v16bf b  = loadFrag(BsF[buf], ni,  lane);
        acc0 = __builtin_amdgcn_wmma_f32_16x16x32_bf16(false, a0, false, b, (short)0, acc0, false, false);
        acc1 = __builtin_amdgcn_wmma_f32_16x16x32_bf16(false, a1, false, b, (short)0, acc1, false, false);
    };

    stage(0);
    const int nk = K / TILE_K;
    for (int it = 0; it < nk - 1; ++it) {
        __syncthreads();
        stage((it + 1) & 1);
        compute(it & 1);
    }
    __syncthreads();
    compute((nk - 1) & 1);

    // ---- epilogue ----
    const int nl = ni * 16 + (lane & 15);
    const int mb = (lane >> 4) << 3;
    const int ng = n0 + nl;
#pragma unroll
    for (int r = 0; r < 8; ++r) {
        int mg0 = m0 + mi0 * 16 + mb + r;
        int mg1 = m0 + mi1 * 16 + mb + r;
        float v0 = acc0[r] + bias[mg0];
        float v1 = acc1[r] + bias[mg1];
        if (EPMODE == EP_BF16_BIAS_GELU) { v0 = gelu_exact(v0); v1 = gelu_exact(v1); }
        size_t o0 = (size_t)bz * M * N + (size_t)mg0 * N + ng;
        size_t o1 = (size_t)bz * M * N + (size_t)mg1 * N + ng;
        if (EPMODE == EP_F32_BIAS) {
            ((float*)Out)[o0] = v0;
            ((float*)Out)[o1] = v1;
        } else {
            ((__bf16*)Out)[o0] = (__bf16)v0;
            ((__bf16*)Out)[o1] = (__bf16)v1;
        }
    }
}

// ---------------------------------------------------------------------------
// Per-(b,c) stats over spatial; optionally emits fused affine-norm coeffs.
// ---------------------------------------------------------------------------
__global__ __launch_bounds__(256)
void stats_kernel(const float* __restrict__ x, const float* __restrict__ gamma,
                  const float* __restrict__ beta, float* __restrict__ out0,
                  float* __restrict__ out1, int S, int C)
{
    const int bc = blockIdx.x;
    const float* p = x + (size_t)bc * S;
    const int lane = threadIdx.x & 31, wave = threadIdx.x >> 5;
    float s = 0.f, s2 = 0.f;
    for (int i = threadIdx.x; i < S; i += 256) { float v = p[i]; s += v; s2 += v * v; }
#pragma unroll
    for (int off = 16; off; off >>= 1) {
        s  += __shfl_xor(s,  off, 32);
        s2 += __shfl_xor(s2, off, 32);
    }
    __shared__ float w1[8], w2[8];
    if (lane == 0) { w1[wave] = s; w2[wave] = s2; }
    __syncthreads();
    if (threadIdx.x == 0) {
        float a = 0.f, b = 0.f;
#pragma unroll
        for (int i = 0; i < 8; ++i) { a += w1[i]; b += w2[i]; }
        float m = a / (float)S;
        float r = rsqrtf(b / (float)S - m * m + 1e-5f);
        if (gamma) {
            int c = bc % C;
            float sc = r * gamma[c];
            out0[bc] = sc;
            out1[bc] = beta[c] - m * sc;
        } else {
            out0[bc] = m;
            out1[bc] = r;
        }
    }
}

// ---------------------------------------------------------------------------
// LayerNorm of q, k over head_dim (64), in place on bf16 qkv. Wave per token.
// ---------------------------------------------------------------------------
__global__ __launch_bounds__(256)
void lnqk_kernel(__bf16* __restrict__ qkv,
                 const float* __restrict__ qw, const float* __restrict__ qb,
                 const float* __restrict__ kw, const float* __restrict__ kb)
{
    const int lane = threadIdx.x & 31, wave = threadIdx.x >> 5;
    size_t gt = (size_t)blockIdx.x * 8 + wave;
    int s = (int)(gt & 4095);
    size_t bh = gt >> 12;
    int he = (int)(bh % NHEADS);
    int b  = (int)(bh / NHEADS);
    size_t base = ((size_t)b * (3 * CEMB) + (size_t)he * 192) * SPAT + s;
#pragma unroll
    for (int part = 0; part < 2; ++part) {
        const float* gw = part ? kw : qw;
        const float* gb = part ? kb : qb;
        size_t pb = base + (size_t)part * 64 * SPAT;
        float v0 = (float)qkv[pb + (size_t)lane * SPAT];
        float v1 = (float)qkv[pb + (size_t)(lane + 32) * SPAT];
        float sm = v0 + v1;
#pragma unroll
        for (int off = 16; off; off >>= 1) sm += __shfl_xor(sm, off, 32);
        float m = sm * (1.f / 64.f);
        float d0 = v0 - m, d1 = v1 - m;
        float sq = d0 * d0 + d1 * d1;
#pragma unroll
        for (int off = 16; off; off >>= 1) sq += __shfl_xor(sq, off, 32);
        float r = rsqrtf(sq * (1.f / 64.f) + 1e-5f);
        qkv[pb + (size_t)lane * SPAT]        = (__bf16)(d0 * r * gw[lane] + gb[lane]);
        qkv[pb + (size_t)(lane + 32) * SPAT] = (__bf16)(d1 * r * gw[lane + 32] + gb[lane + 32]);
    }
}

// ---------------------------------------------------------------------------
// Axial attention (one block per b,head,line), WMMA for QK^T and P@V.
// Per-head rel-bias row staged via global_load_async_to_lds (ASYNCcnt).
// AXIS=1 fuses xa = 0.5*(xx+xy) and the transpose to (b, c, s).
// ---------------------------------------------------------------------------
template<int AXIS>
__global__ __launch_bounds__(256)
void axial_attn_kernel(const __bf16* __restrict__ qkv,
                       const float* __restrict__ rel_table,
                       const float* __restrict__ scale_arr,
                       const float* __restrict__ xx_in,
                       float* __restrict__ out)
{
    const int line = blockIdx.x, he = blockIdx.y, b = blockIdx.z;
    const int tid = threadIdx.x, lane = tid & 31, wave = tid >> 5;

    const size_t cq = ((size_t)b * (3 * CEMB) + (size_t)he * 192) * SPAT;
    const int sbase = (AXIS == 0) ? line * SIDE : line;
    const int sstr  = (AXIS == 0) ? 1 : SIDE;

    __shared__ __align__(32) __bf16 QsF[2][2048];   // A-frag: m=token, k=d
    __shared__ __align__(32) __bf16 KsF[2][2048];   // B-frag: n=token, k=d
    __shared__ __align__(32) __bf16 VsF[2][2048];   // B-frag: n=d, k=token
    __shared__ __align__(32) __bf16 PsF[2][2048];   // A-frag: m=row,  k=token
    __shared__ float Sf[64][65];
    __shared__ float biasLds[128];

    // ---- async-copy the per-head relative-bias row (127 floats) to LDS ----
    if (tid < 127) {
        unsigned loff = (unsigned)(uintptr_t)&biasLds[tid];
        unsigned long long ga =
            (unsigned long long)(uintptr_t)(rel_table + (size_t)tid * NHEADS + he);
        asm volatile("global_load_async_to_lds_b32 %0, %1, off"
                     :: "v"(loff), "v"(ga) : "memory");
    }

    // ---- stage Q, K (thread: fixed token, 4 consecutive d) ----
#pragma unroll
    for (int p = 0; p < 4; ++p) {
        int tt = tid & 63;
        int dq = ((tid >> 6) + p * 4) * 4;
        size_t a = cq + (size_t)dq * SPAT + (size_t)(sbase + tt * sstr);
        float q0 = (float)qkv[a];
        float q1 = (float)qkv[a + SPAT];
        float q2 = (float)qkv[a + 2 * SPAT];
        float q3 = (float)qkv[a + 3 * SPAT];
        size_t ak = a + (size_t)64 * SPAT;
        float k0 = (float)qkv[ak];
        float k1 = (float)qkv[ak + SPAT];
        float k2 = (float)qkv[ak + 2 * SPAT];
        float k3 = (float)qkv[ak + 3 * SPAT];
        *(v4bf*)(QsF[dq >> 5] + fragAoff(tt, dq & 31)) = pack4(q0, q1, q2, q3);
        *(v4bf*)(KsF[dq >> 5] + fragBoff(tt, dq & 31)) = pack4(k0, k1, k2, k3);
    }
    // ---- stage V (thread: fixed d, 4 consecutive tokens) ----
#pragma unroll
    for (int p = 0; p < 4; ++p) {
        int d  = tid & 63;
        int tq = ((tid >> 6) + p * 4) * 4;
        size_t a = cq + (size_t)(128 + d) * SPAT + (size_t)(sbase + tq * sstr);
        float v0 = (float)qkv[a];
        float v1 = (float)qkv[a + (size_t)sstr];
        float v2 = (float)qkv[a + (size_t)2 * sstr];
        float v3 = (float)qkv[a + (size_t)3 * sstr];
        *(v4bf*)(VsF[tq >> 5] + fragBoff(d, tq & 31)) = pack4(v0, v1, v2, v3);
    }
    asm volatile("s_wait_asynccnt 0x0" ::: "memory");
    __syncthreads();

    const int mi0 = wave >> 2, ni = wave & 3, mi1 = mi0 + 2;
    const int nl = ni * 16 + (lane & 15);
    const int mb = (lane >> 4) << 3;

    // ---- GEMM1: S = Q @ K^T ----
    v8f s0 = {}, s1 = {};
#pragma unroll
    for (int ks = 0; ks < 2; ++ks) {
        v16bf a0 = loadFrag(QsF[ks], mi0, lane);
        v16bf a1 = loadFrag(QsF[ks], mi1, lane);
        v16bf bb = loadFrag(KsF[ks], ni,  lane);
        s0 = __builtin_amdgcn_wmma_f32_16x16x32_bf16(false, a0, false, bb, (short)0, s0, false, false);
        s1 = __builtin_amdgcn_wmma_f32_16x16x32_bf16(false, a1, false, bb, (short)0, s1, false, false);
    }

    const float scaling = 0.125f;
#pragma unroll
    for (int r = 0; r < 8; ++r) {
        int q0 = mi0 * 16 + mb + r;
        int q1 = mi1 * 16 + mb + r;
        Sf[q0][nl] = s0[r] * scaling + biasLds[q0 - nl + 63];
        Sf[q1][nl] = s1[r] * scaling + biasLds[q1 - nl + 63];
    }
    __syncthreads();

    // ---- softmax + low-mix -> P (A-frag layout) ----
    const float scv = scale_arr[he];
    const float low = 1.f / 64.f;
    for (int row = wave; row < 64; row += 8) {
        float x0 = Sf[row][lane], x1 = Sf[row][lane + 32];
        float mx = fmaxf(x0, x1);
#pragma unroll
        for (int off = 16; off; off >>= 1) mx = fmaxf(mx, __shfl_xor(mx, off, 32));
        float e0 = __expf(x0 - mx), e1 = __expf(x1 - mx);
        float sm = e0 + e1;
#pragma unroll
        for (int off = 16; off; off >>= 1) sm += __shfl_xor(sm, off, 32);
        float inv = 1.f / sm;
        PsF[0][fragAoff(row, lane)] = (__bf16)(low + (e0 * inv - low) * scv);
        PsF[1][fragAoff(row, lane)] = (__bf16)(low + (e1 * inv - low) * scv);
    }
    __syncthreads();

    // ---- GEMM2: O = P @ V ----
    v8f o0 = {}, o1 = {};
#pragma unroll
    for (int ks = 0; ks < 2; ++ks) {
        v16bf a0 = loadFrag(PsF[ks], mi0, lane);
        v16bf a1 = loadFrag(PsF[ks], mi1, lane);
        v16bf bb = loadFrag(VsF[ks], ni,  lane);
        o0 = __builtin_amdgcn_wmma_f32_16x16x32_bf16(false, a0, false, bb, (short)0, o0, false, false);
        o1 = __builtin_amdgcn_wmma_f32_16x16x32_bf16(false, a1, false, bb, (short)0, o1, false, false);
    }

    // ---- epilogue ----
    const size_t hb = (size_t)b * NHEADS + he;
#pragma unroll
    for (int r = 0; r < 8; ++r) {
        int t0 = mi0 * 16 + mb + r;
        int t1 = mi1 * 16 + mb + r;
        if (AXIS == 0) {
            out[(hb * SPAT + (size_t)line * SIDE + t0) * HD + nl] = o0[r];
            out[(hb * SPAT + (size_t)line * SIDE + t1) * HD + nl] = o1[r];
        } else {
            size_t xi0 = (hb * SPAT + (size_t)t0 * SIDE + line) * HD + nl;
            size_t xi1 = (hb * SPAT + (size_t)t1 * SIDE + line) * HD + nl;
            float v0 = 0.5f * (xx_in[xi0] + o0[r]);
            float v1 = 0.5f * (xx_in[xi1] + o1[r]);
            size_t c = (size_t)b * CEMB + (size_t)he * HD + nl;
            out[c * SPAT + (size_t)t0 * SIDE + line] = v0;
            out[c * SPAT + (size_t)t1 * SIDE + line] = v1;
        }
    }
}

// ---------------------------------------------------------------------------
__global__ __launch_bounds__(256)
void featscale_kernel(float* __restrict__ y, const float* __restrict__ ymean,
                      const float* __restrict__ low_s, const float* __restrict__ high_s)
{
    const size_t total = (size_t)NB * CEMB * SPAT;
    for (size_t i = (size_t)blockIdx.x * 256 + threadIdx.x; i < total;
         i += (size_t)gridDim.x * 256) {
        size_t bc = i >> 12;
        int c = (int)(bc % CEMB);
        float xl = ymean[bc];
        float v = y[i];
        y[i] = v + xl * low_s[c] + (v - xl) * high_s[c];
    }
}

__global__ __launch_bounds__(256)
void final_kernel(const float* __restrict__ xm, const float* __restrict__ sc,
                  const float* __restrict__ sh, const float* __restrict__ resid,
                  float* __restrict__ out)
{
    const size_t total = (size_t)NB * CEMB * SPAT;
    for (size_t i = (size_t)blockIdx.x * 256 + threadIdx.x; i < total;
         i += (size_t)gridDim.x * 256) {
        size_t bc = i >> 12;
        out[i] = xm[i] * sc[bc] + sh[bc] + resid[i];
    }
}

// ---------------------------------------------------------------------------
extern "C" void kernel_launch(void* const* d_in, const int* in_sizes, int n_in,
                              void* d_out, int out_size, void* d_ws, size_t ws_size,
                              hipStream_t stream)
{
    const float* x       = (const float*)d_in[0];
    const float* n1_w    = (const float*)d_in[1];
    const float* n1_b    = (const float*)d_in[2];
    const float* w_in    = (const float*)d_in[3];
    const float* b_in    = (const float*)d_in[4];
    const float* q_w     = (const float*)d_in[5];
    const float* q_b     = (const float*)d_in[6];
    const float* k_w     = (const float*)d_in[7];
    const float* k_b     = (const float*)d_in[8];
    const float* rel_t   = (const float*)d_in[9];
    const float* scale_x = (const float*)d_in[10];
    const float* scale_y = (const float*)d_in[11];
    const float* n2_w    = (const float*)d_in[12];
    const float* n2_b    = (const float*)d_in[13];
    const float* w_out   = (const float*)d_in[14];
    const float* b_out   = (const float*)d_in[15];
    const float* low_s   = (const float*)d_in[16];
    const float* high_s  = (const float*)d_in[17];
    const float* mlp_w1  = (const float*)d_in[18];
    const float* mlp_b1  = (const float*)d_in[19];
    const float* mlp_w2  = (const float*)d_in[20];
    const float* mlp_b2  = (const float*)d_in[21];
    const float* nm_w    = (const float*)d_in[22];
    const float* nm_b    = (const float*)d_in[23];

    char* ws = (char*)d_ws;
    const size_t SZ_QKV = (size_t)NB * (3 * CEMB) * SPAT * 2;   // bf16
    const size_t SZ_F   = (size_t)NB * CEMB * SPAT * 4;         // fp32
    const size_t OFF_QKV = 0;
    const size_t OFF_XX  = OFF_QKV + SZ_QKV;
    const size_t OFF_XA  = OFF_XX + SZ_F;
    const size_t OFF_Y   = OFF_XA + SZ_F;       // becomes residual in place
    const size_t OFF_XM  = OFF_Y + SZ_F;
    const size_t OFF_ST  = OFF_XM + SZ_F;

    __bf16* qkv = (__bf16*)(ws + OFF_QKV);
    float*  xx  = (float*)(ws + OFF_XX);
    float*  xa  = (float*)(ws + OFF_XA);
    float*  y   = (float*)(ws + OFF_Y);
    float*  xm  = (float*)(ws + OFF_XM);
    __bf16* h1  = (__bf16*)(ws + OFF_QKV);      // reuses qkv+xx after attention
    float*  st  = (float*)(ws + OFF_ST);
    float *sA1 = st,          *sB1 = st + 6144;
    float *sA2 = st + 12288,  *sB2 = st + 18432;
    float *m3  = st + 24576,  *r3  = st + 30720;
    float *sc4 = st + 36864,  *sh4 = st + 43008;

    const dim3 blk(256);

    // 1) instance-norm coefficients for x
    stats_kernel<<<dim3(NB * CEMB), blk, 0, stream>>>(x, n1_w, n1_b, sA1, sB1, SPAT, CEMB);

    // 2) qkv = w_in @ inorm(x) + b_in -> bf16 [b][2304][4096]
    gemm_wmma_kernel<0, BMODE_F32_NORM, EP_BF16_BIAS>
        <<<dim3(SPAT / TILE_N, (3 * CEMB) / TILE_M, NB), blk, 0, stream>>>(
        w_in, CEMB, x, sA1, sB1, b_in, qkv, 3 * CEMB, SPAT, CEMB);

    // 3) layernorm q, k over head_dim
    lnqk_kernel<<<dim3(NB * NHEADS * SPAT / 8), blk, 0, stream>>>(qkv, q_w, q_b, k_w, k_b);

    // 4) attention along W -> xx [b][he][s][d]
    axial_attn_kernel<0><<<dim3(SIDE, NHEADS, NB), blk, 0, stream>>>(
        qkv, rel_t, scale_x, nullptr, xx);

    // 5) attention along H, fused xa = 0.5*(xx+xy) -> xa [b][c][s]
    axial_attn_kernel<1><<<dim3(SIDE, NHEADS, NB), blk, 0, stream>>>(
        qkv, rel_t, scale_y, xx, xa);

    // 6) instance-norm coefficients for xa
    stats_kernel<<<dim3(NB * CEMB), blk, 0, stream>>>(xa, n2_w, n2_b, sA2, sB2, SPAT, CEMB);

    // 7) y = w_out @ inorm(xa) + b_out -> fp32
    gemm_wmma_kernel<0, BMODE_F32_NORM, EP_F32_BIAS>
        <<<dim3(SPAT / TILE_N, CEMB / TILE_M, NB), blk, 0, stream>>>(
        w_out, CEMB, xa, sA2, sB2, b_out, y, CEMB, SPAT, CEMB);

    // 8) spatial mean of y (raw stats)
    stats_kernel<<<dim3(NB * CEMB), blk, 0, stream>>>(y, nullptr, nullptr, m3, r3, SPAT, CEMB);

    // 9) feat_scale in place; y becomes residual
    featscale_kernel<<<dim3(4096), blk, 0, stream>>>(y, m3, low_s, high_s);

    // 10) h1 = gelu(W1^T @ resid + b1) -> bf16 [b][3072][4096]
    gemm_wmma_kernel<1, BMODE_F32, EP_BF16_BIAS_GELU>
        <<<dim3(SPAT / TILE_N, (4 * CEMB) / TILE_M, NB), blk, 0, stream>>>(
        mlp_w1, 4 * CEMB, y, nullptr, nullptr, mlp_b1, h1, 4 * CEMB, SPAT, CEMB);

    // 11) xm = W2^T @ h1 + b2 -> fp32 [b][768][4096]
    gemm_wmma_kernel<1, BMODE_BF16, EP_F32_BIAS>
        <<<dim3(SPAT / TILE_N, CEMB / TILE_M, NB), blk, 0, stream>>>(
        mlp_w2, CEMB, h1, nullptr, nullptr, mlp_b2, xm, CEMB, SPAT, 4 * CEMB);

    // 12) fused inorm coefficients for xm
    stats_kernel<<<dim3(NB * CEMB), blk, 0, stream>>>(xm, nm_w, nm_b, sc4, sh4, SPAT, CEMB);

    // 13) out = xm*sc + sh + resid
    final_kernel<<<dim3(4096), blk, 0, stream>>>(xm, sc4, sh4, y, (float*)d_out);
}
